// GraphModel_57131654971883
// MI455X (gfx1250) — matrix operand order repaired and checked by reading, not compiled, not run on hardware
//
#include <hip/hip_runtime.h>

// ---------------------------------------------------------------------------
// Types matching gfx1250 WMMA operand layouts (wave32)
// ---------------------------------------------------------------------------
typedef __attribute__((ext_vector_type(16))) __bf16          v16bf;
typedef __attribute__((ext_vector_type(8)))  float           v8f;
typedef __attribute__((ext_vector_type(8)))  unsigned short  v8us;
typedef __attribute__((ext_vector_type(4)))  float           v4f;

union BF16x16 { v8us h[2]; v16bf v; };

__device__ __forceinline__ unsigned short f2bf(float f) {
    unsigned int u = __float_as_uint(f);
    u += 0x7FFFu + ((u >> 16) & 1u);           // round-to-nearest-even
    return (unsigned short)(u >> 16);
}

// LDS byte offset of a __shared__ object (generic LDS ptr: low 32 bits = offset)
__device__ __forceinline__ unsigned lds_off(const void* p) {
    return (unsigned)(unsigned long long)p;
}

// Async global -> LDS 16-byte copy (CDNA5, tracked by ASYNCcnt)
__device__ __forceinline__ void async_copy_b128(unsigned dst_lds, const void* g) {
    asm volatile("global_load_async_to_lds_b128 %0, %1, off"
                 :: "v"(dst_lds), "v"((unsigned long long)g)
                 : "memory");
}
__device__ __forceinline__ void wait_async() {
    asm volatile("s_wait_asynccnt 0" ::: "memory");
}

#define NFEAT   256
#define N_NODES 50000
#define MTILE   64          // edges per block
#define LRELU(x) ((x) > 0.f ? (x) : 0.01f * (x))

// ---------------------------------------------------------------------------
// Weight transpose + f32->bf16 convert: W[K][N] -> Wt[N][K] (bf16)
// ---------------------------------------------------------------------------
__global__ void wt_convert_kernel(const float* __restrict__ W,
                                  unsigned short* __restrict__ Wt,
                                  int K, int N) {
    int i = blockIdx.x * blockDim.x + threadIdx.x;
    if (i >= K * N) return;
    int n = i % N, k = i / N;
    Wt[(long long)n * K + k] = f2bf(W[i]);
}

// ---------------------------------------------------------------------------
// In-degree counts (float, for mean divide)
// ---------------------------------------------------------------------------
__global__ void count_kernel(const int* __restrict__ d0, const int* __restrict__ d1,
                             float* __restrict__ c0, float* __restrict__ c1, int E) {
    int i = blockIdx.x * blockDim.x + threadIdx.x;
    if (i >= E) return;
    atomicAdd(&c0[d0[i]], 1.0f);
    atomicAdd(&c1[d1[i]], 1.0f);
}

// ---------------------------------------------------------------------------
// Phase A: m = leaky_relu(e @ W + b); relsum[dst] += m  (WMMA bf16 GEMM)
// Block: 256 thr (8 waves), 64 edges, N=256, K=256.
// Wave w owns N columns [w*32, w*32+32); B frags reused across 4 M-subtiles.
// ---------------------------------------------------------------------------
#define APAD 264   // 256 + 8 bf16 pad -> 16B-aligned rows, 4-bank rotation

__global__ __launch_bounds__(256)
void msg_kernel(const float* __restrict__ e, const int* __restrict__ dst,
                const unsigned short* __restrict__ Wt, const float* __restrict__ bias,
                float* __restrict__ relsum, int E) {
    __shared__ __align__(16) unsigned short sA[MTILE * APAD];
    __shared__ int sDst[MTILE];
    const int t = threadIdx.x;
    const long long ebase = (long long)blockIdx.x * MTILE;

    if (t < MTILE) {
        long long er = ebase + t;
        sDst[t] = dst[er < E ? er : (E - 1)];
    }
    // cooperative fill: 64 rows x 256 f32 -> bf16 LDS (4 chunks of 16 / thread)
#pragma unroll
    for (int c = 0; c < 4; ++c) {
        const int idx = (t + c * 256) * 16;
        const int row = idx >> 8, col = idx & 255;
        long long er = ebase + row; if (er >= E) er = E - 1;
        const float* sp = e + er * NFEAT + col;
        unsigned short tmp[16];
#pragma unroll
        for (int j = 0; j < 4; ++j) {
            v4f f = *(const v4f*)(sp + 4 * j);
            tmp[4*j+0] = f2bf(f.x); tmp[4*j+1] = f2bf(f.y);
            tmp[4*j+2] = f2bf(f.z); tmp[4*j+3] = f2bf(f.w);
        }
        *(v8us*)&sA[row * APAD + col]     = *(v8us*)&tmp[0];
        *(v8us*)&sA[row * APAD + col + 8] = *(v8us*)&tmp[8];
    }
    __syncthreads();

    const int wave = t >> 5, lane = t & 31;
    const int kh = lane >> 4;        // half-wave index
    const int lr = lane & 15;
    const int nb = wave * 32;

    v8f acc[4][2] = {};
#pragma unroll
    for (int ks = 0; ks < 8; ++ks) {
        BF16x16 b0, b1;
        // B: lane<16 holds col n, K 0-15 of step; lane>=16: K 16-31 (Wt is [N][K])
        const unsigned short* bp0 = Wt + (long long)(nb + lr) * NFEAT + ks * 32 + 16 * kh;
        const unsigned short* bp1 = bp0 + 16 * NFEAT;
        b0.h[0] = *(const v8us*)bp0;  b0.h[1] = *(const v8us*)(bp0 + 8);
        b1.h[0] = *(const v8us*)bp1;  b1.h[1] = *(const v8us*)(bp1 + 8);
#pragma unroll
        for (int mi = 0; mi < 4; ++mi) {
            BF16x16 a;
            // A: lane<16 holds row lr, K 0-15; lane>=16: K 16-31
            const unsigned short* ap = &sA[(mi * 16 + lr) * APAD + ks * 32 + 16 * kh];
            a.h[0] = *(const v8us*)ap;
            a.h[1] = *(const v8us*)(ap + 8);
            acc[mi][0] = __builtin_amdgcn_wmma_f32_16x16x32_bf16(
                            false, a.v, false, b0.v, (short)0, acc[mi][0], false, false);
            acc[mi][1] = __builtin_amdgcn_wmma_f32_16x16x32_bf16(
                            false, a.v, false, b1.v, (short)0, acc[mi][1], false, false);
        }
    }

    const float bn0 = bias[nb + lr];
    const float bn1 = bias[nb + 16 + lr];
#pragma unroll
    for (int mi = 0; mi < 4; ++mi) {
#pragma unroll
        for (int v = 0; v < 8; ++v) {
            const int m = mi * 16 + 8 * kh + v;        // edge row in tile
            if (ebase + m < E) {
                const long long rb = (long long)sDst[m] * NFEAT;
                float x0 = LRELU(acc[mi][0][v] + bn0);
                float x1 = LRELU(acc[mi][1][v] + bn1);
                atomicAdd(&relsum[rb + nb + lr], x0);
                atomicAdd(&relsum[rb + nb + 16 + lr], x1);
            }
        }
    }
}

// ---------------------------------------------------------------------------
// rel = sum / max(count,1) -> bf16
// ---------------------------------------------------------------------------
__global__ void mean_kernel(const float* __restrict__ s, const float* __restrict__ c,
                            unsigned short* __restrict__ out, long long total) {
    long long i = (long long)blockIdx.x * blockDim.x + threadIdx.x;
    if (i >= total) return;
    float cnt = fmaxf(c[i >> 8], 1.0f);
    out[i] = f2bf(s[i] / cnt);
}

// ---------------------------------------------------------------------------
// Phase B: e' = [e | rel0[src] | rel1[dst]] @ We + be   (K = 768)
// 64 edges/block; rel gathers use async global->LDS (ASYNCcnt) copies.
// ---------------------------------------------------------------------------
#define BPAD 776   // 768 + 8 bf16 pad

__global__ __launch_bounds__(256)
void edge_kernel(const float* __restrict__ e,
                 const int* __restrict__ src, const int* __restrict__ dst,
                 const unsigned short* __restrict__ rel0,
                 const unsigned short* __restrict__ rel1,
                 const unsigned short* __restrict__ Wt,   // [256][768] bf16
                 const float* __restrict__ bias,
                 float* __restrict__ out, int E) {
    __shared__ __align__(16) unsigned short sA[MTILE * BPAD];
    __shared__ int sSrc[MTILE], sDst[MTILE];
    const int t = threadIdx.x;
    const long long ebase = (long long)blockIdx.x * MTILE;

    if (t < MTILE) {
        long long er = ebase + t;
        long long ec = er < E ? er : (E - 1);
        sSrc[t] = src[ec];
        sDst[t] = dst[ec];
    }
    __syncthreads();

#pragma unroll
    for (int c = 0; c < 4; ++c) {
        const int idx = (t + c * 256) * 16;
        const int row = idx >> 8, col = idx & 255;
        // segment 0: e (f32 -> bf16, VALU convert)
        {
            long long er = ebase + row; if (er >= E) er = E - 1;
            const float* sp = e + er * NFEAT + col;
            unsigned short tmp[16];
#pragma unroll
            for (int j = 0; j < 4; ++j) {
                v4f f = *(const v4f*)(sp + 4 * j);
                tmp[4*j+0] = f2bf(f.x); tmp[4*j+1] = f2bf(f.y);
                tmp[4*j+2] = f2bf(f.z); tmp[4*j+3] = f2bf(f.w);
            }
            *(v8us*)&sA[row * BPAD + col]     = *(v8us*)&tmp[0];
            *(v8us*)&sA[row * BPAD + col + 8] = *(v8us*)&tmp[8];
        }
        // segment 1: rel0[src[row]]  (async global->LDS copy, bf16)
        {
            const unsigned short* p = rel0 + (long long)sSrc[row] * NFEAT + col;
            async_copy_b128(lds_off(&sA[row * BPAD + 256 + col]), p);
            async_copy_b128(lds_off(&sA[row * BPAD + 256 + col + 8]), p + 8);
        }
        // segment 2: rel1[dst[row]]
        {
            const unsigned short* p = rel1 + (long long)sDst[row] * NFEAT + col;
            async_copy_b128(lds_off(&sA[row * BPAD + 512 + col]), p);
            async_copy_b128(lds_off(&sA[row * BPAD + 512 + col + 8]), p + 8);
        }
    }
    wait_async();
    __syncthreads();

    const int wave = t >> 5, lane = t & 31;
    const int kh = lane >> 4, lr = lane & 15;
    const int nb = wave * 32;

    v8f acc[4][2] = {};
    for (int ks = 0; ks < 24; ++ks) {
        BF16x16 b0, b1;
        const unsigned short* bp0 = Wt + (long long)(nb + lr) * 768 + ks * 32 + 16 * kh;
        const unsigned short* bp1 = bp0 + 16 * 768;
        b0.h[0] = *(const v8us*)bp0;  b0.h[1] = *(const v8us*)(bp0 + 8);
        b1.h[0] = *(const v8us*)bp1;  b1.h[1] = *(const v8us*)(bp1 + 8);
#pragma unroll
        for (int mi = 0; mi < 4; ++mi) {
            BF16x16 a;
            const unsigned short* ap = &sA[(mi * 16 + lr) * BPAD + ks * 32 + 16 * kh];
            a.h[0] = *(const v8us*)ap;
            a.h[1] = *(const v8us*)(ap + 8);
            acc[mi][0] = __builtin_amdgcn_wmma_f32_16x16x32_bf16(
                            false, a.v, false, b0.v, (short)0, acc[mi][0], false, false);
            acc[mi][1] = __builtin_amdgcn_wmma_f32_16x16x32_bf16(
                            false, a.v, false, b1.v, (short)0, acc[mi][1], false, false);
        }
    }

    const float bn0 = bias[nb + lr];
    const float bn1 = bias[nb + 16 + lr];
#pragma unroll
    for (int mi = 0; mi < 4; ++mi) {
#pragma unroll
        for (int v = 0; v < 8; ++v) {
            const int m = mi * 16 + 8 * kh + v;
            if (ebase + m < E) {
                float* op = out + (ebase + m) * NFEAT;
                op[nb + lr]      = acc[mi][0][v] + bn0;
                op[nb + 16 + lr] = acc[mi][1][v] + bn1;
            }
        }
    }
}

// ---------------------------------------------------------------------------
// Host launcher
// ---------------------------------------------------------------------------
extern "C" void kernel_launch(void* const* d_in, const int* in_sizes, int n_in,
                              void* d_out, int out_size, void* d_ws, size_t ws_size,
                              hipStream_t stream) {
    const float* e0  = (const float*)d_in[0];
    const float* e1  = (const float*)d_in[1];
    const int* src0  = (const int*)d_in[2];
    const int* dst0  = (const int*)d_in[3];
    const int* src1  = (const int*)d_in[4];
    const int* dst1  = (const int*)d_in[5];
    const float* W0  = (const float*)d_in[6];
    const float* b0  = (const float*)d_in[7];
    const float* W1  = (const float*)d_in[8];
    const float* b1  = (const float*)d_in[9];
    const float* We0 = (const float*)d_in[10];
    const float* be0 = (const float*)d_in[11];
    const float* We1 = (const float*)d_in[12];
    const float* be1 = (const float*)d_in[13];

    const int E = in_sizes[0] / NFEAT;          // 150000
    const long long relElems = (long long)N_NODES * NFEAT;

    // ---- workspace carve (256B aligned slots) ----
    char* ws = (char*)d_ws;
    size_t off = 0;
    auto carve = [&](size_t bytes) -> void* {
        void* p = ws + off;
        off += (bytes + 255) & ~(size_t)255;
        return p;
    };
    float* relsum0 = (float*)carve(relElems * 4);
    float* relsum1 = (float*)carve(relElems * 4);
    float* cnt0    = (float*)carve((size_t)N_NODES * 4);
    float* cnt1    = (float*)carve((size_t)N_NODES * 4);
    unsigned short* relbf0 = (unsigned short*)carve(relElems * 2);
    unsigned short* relbf1 = (unsigned short*)carve(relElems * 2);
    unsigned short* Wt0  = (unsigned short*)carve((size_t)256 * 256 * 2);
    unsigned short* Wt1  = (unsigned short*)carve((size_t)256 * 256 * 2);
    unsigned short* Wet0 = (unsigned short*)carve((size_t)256 * 768 * 2);
    unsigned short* Wet1 = (unsigned short*)carve((size_t)256 * 768 * 2);

    float* out0 = (float*)d_out;
    float* out1 = out0 + (long long)E * NFEAT;

    // ---- zero accumulators (every call: deterministic) ----
    hipMemsetAsync(relsum0, 0, relElems * 4, stream);
    hipMemsetAsync(relsum1, 0, relElems * 4, stream);
    hipMemsetAsync(cnt0,    0, (size_t)N_NODES * 4, stream);
    hipMemsetAsync(cnt1,    0, (size_t)N_NODES * 4, stream);

    // ---- weight transpose/convert ----
    {
        int n256 = 256 * 256, n768 = 768 * 256;
        wt_convert_kernel<<<(n256 + 255) / 256, 256, 0, stream>>>(W0,  Wt0,  256, 256);
        wt_convert_kernel<<<(n256 + 255) / 256, 256, 0, stream>>>(W1,  Wt1,  256, 256);
        wt_convert_kernel<<<(n768 + 255) / 256, 256, 0, stream>>>(We0, Wet0, 768, 256);
        wt_convert_kernel<<<(n768 + 255) / 256, 256, 0, stream>>>(We1, Wet1, 768, 256);
    }

    // ---- degree counts ----
    count_kernel<<<(E + 255) / 256, 256, 0, stream>>>(dst0, dst1, cnt0, cnt1, E);

    // ---- phase A: message GEMM + leaky_relu + segment-sum ----
    const int nTiles = (E + MTILE - 1) / MTILE;
    msg_kernel<<<nTiles, 256, 0, stream>>>(e0, dst0, Wt0, b0, relsum0, E);
    msg_kernel<<<nTiles, 256, 0, stream>>>(e1, dst1, Wt1, b1, relsum1, E);

    // ---- mean + bf16 convert ----
    mean_kernel<<<(unsigned)((relElems + 255) / 256), 256, 0, stream>>>(relsum0, cnt0, relbf0, relElems);
    mean_kernel<<<(unsigned)((relElems + 255) / 256), 256, 0, stream>>>(relsum1, cnt1, relbf1, relElems);

    // ---- phase B: edge-update GEMM on gathered concat ----
    edge_kernel<<<nTiles, 256, 0, stream>>>(e0, src0, dst0, relbf0, relbf1, Wet0, be0, out0, E);
    edge_kernel<<<nTiles, 256, 0, stream>>>(e1, src1, dst1, relbf0, relbf1, Wet1, be1, out1, E);
}
